// Attention_54382875902423
// MI455X (gfx1250) — matrix operand order repaired and checked by reading, not compiled, not run on hardware
//
#include <hip/hip_runtime.h>
#include <hip/hip_bf16.h>

typedef __attribute__((ext_vector_type(16))) _Float16 v16h;
typedef __attribute__((ext_vector_type(4)))  _Float16 v4h;
typedef __attribute__((ext_vector_type(8)))  float    v8f;
typedef int v4i __attribute__((__vector_size__(4 * sizeof(int))));

#define B_   4
#define S_   1024
#define D_   4096
#define H_   32
#define HD_  128
#define NK   4096
#define NELEM ((size_t)4096 * 4096)

#define AS1 __attribute__((address_space(1)))
#define AS3 __attribute__((address_space(3)))

// ---- CDNA5 async global->LDS (ASYNCcnt) with sync fallback -----------------
#if defined(__has_builtin)
#if __has_builtin(__builtin_amdgcn_global_load_async_to_lds_b128)
#define ASYNC_LDS 1
#endif
#endif
#ifndef ASYNC_LDS
#define ASYNC_LDS 0
#endif

__device__ __forceinline__ void copy16_g2l(const _Float16* g, _Float16* l) {
#if ASYNC_LDS
    __builtin_amdgcn_global_load_async_to_lds_b128(
        (AS1 v4i*)g, (AS3 v4i*)l, 0, 0);
#else
    *(uint4*)l = *(const uint4*)g;
#endif
}

__device__ __forceinline__ void async_wait_all() {
#if ASYNC_LDS
#if defined(__has_builtin) && __has_builtin(__builtin_amdgcn_s_wait_asynccnt)
    __builtin_amdgcn_s_wait_asynccnt(0);
#else
    asm volatile("s_wait_asynccnt 0x0" ::: "memory");
#endif
#endif
}

// ---------------------------------------------------------------------------
// fp32 -> f16 straight cast (x)
// ---------------------------------------------------------------------------
__global__ void cast_f32_to_f16_kernel(const float* __restrict__ src,
                                       _Float16* __restrict__ dst, int n4) {
    int i = blockIdx.x * blockDim.x + threadIdx.x;
    if (i < n4) {
        float4 f = ((const float4*)src)[i];
        v4h h;
        h[0] = (_Float16)f.x; h[1] = (_Float16)f.y;
        h[2] = (_Float16)f.z; h[3] = (_Float16)f.w;
        ((v4h*)dst)[i] = h;
    }
}

// ---------------------------------------------------------------------------
// fp32 [K][N] -> f16 [N][K] tiled transpose-cast (weights)
// ---------------------------------------------------------------------------
__global__ __launch_bounds__(256) void transpose_cast_kernel(
    const float* __restrict__ src, _Float16* __restrict__ dst) {
    __shared__ _Float16 t[32][33];
    int tx = threadIdx.x & 31, ty = threadIdx.x >> 5;      // ty 0..7
    int k0 = blockIdx.y * 32, n0 = blockIdx.x * 32;
    #pragma unroll
    for (int i = 0; i < 4; ++i) {
        int r = ty + i * 8;
        t[r][tx] = (_Float16)src[(size_t)(k0 + r) * 4096 + n0 + tx];
    }
    __syncthreads();
    #pragma unroll
    for (int i = 0; i < 4; ++i) {
        int r = ty + i * 8;
        dst[(size_t)(n0 + r) * 4096 + k0 + tx] = t[tx][r];
    }
}

// ---------------------------------------------------------------------------
// WMMA GEMM: C[4096][4096] = A[M][K] * Bt[N][K]^T + bias[N]
// 128x128 block tile, 256 threads = 8 waves (4x2), each wave 32x64 (8 WMMA).
// Double-buffered async LDS staging; all LDS reads are aligned b128.
//   mode 0: f16 out scattered to [B,H,S,HD]     (Q,K projections)
//   mode 2: f16 out scattered to [B,H,HD,S]     (V projection, pre-transposed)
//   mode 1: f32 out row-major [M][N]            (output projection -> d_out)
// ---------------------------------------------------------------------------
__global__ __launch_bounds__(256) void gemm_wmma_kernel(
    const _Float16* __restrict__ A,      // [M][K]
    const _Float16* __restrict__ Bt,     // [N][K]
    const float*    __restrict__ bias,   // [N]
    _Float16* __restrict__ outH,
    float*    __restrict__ outF,
    int mode) {
    __shared__ _Float16 Al[2][128][40];
    __shared__ _Float16 Bl[2][128][40];

    const int tid  = threadIdx.x;
    const int lane = tid & 31;
    const int wave = tid >> 5;
    const int wm   = wave >> 1;          // 0..3 -> rows wm*32
    const int wn   = wave & 1;           // 0..1 -> cols wn*64
    const int lr   = lane & 15;
    const int lh   = lane >> 4;
    const int m0   = blockIdx.y * 128;
    const int n0   = blockIdx.x * 128;

    v8f acc[2][4];
    #pragma unroll
    for (int f = 0; f < 2; ++f)
        #pragma unroll
        for (int g = 0; g < 4; ++g)
            #pragma unroll
            for (int e = 0; e < 8; ++e) acc[f][g][e] = 0.0f;

    // stage one 128x32 K-slab of A and of Bt into LDS buffer `b`
    auto stage = [&](int b, int k0) {
        #pragma unroll
        for (int u = 0; u < 2; ++u) {
            int j   = tid * 2 + u;                 // 0..511
            int row = j >> 2;                      // 0..127
            int c4  = (j & 3) * 8;                 // 0,8,16,24
            copy16_g2l(&A [(size_t)(m0 + row) * NK + k0 + c4], &Al[b][row][c4]);
            copy16_g2l(&Bt[(size_t)(n0 + row) * NK + k0 + c4], &Bl[b][row][c4]);
        }
    };

    stage(0, 0);
    for (int k0 = 0, buf = 0; k0 < NK; k0 += 32, buf ^= 1) {
        async_wait_all();
        __syncthreads();                           // slab ready; prev reads done
        if (k0 + 32 < NK) stage(buf ^ 1, k0 + 32);

        v16h af[2], bf[4];
        #pragma unroll
        for (int f = 0; f < 2; ++f) {
            int row = wm * 32 + f * 16 + lr;
            int kb  = lh * 8;
            ((uint4*)&af[f])[0] = *(const uint4*)&Al[buf][row][kb];
            ((uint4*)&af[f])[1] = *(const uint4*)&Al[buf][row][kb + 16];
        }
        #pragma unroll
        for (int g = 0; g < 4; ++g) {
            int col = wn * 64 + g * 16 + lr;
            int kb2 = lh * 8;
            ((uint4*)&bf[g])[0] = *(const uint4*)&Bl[buf][col][kb2];
            ((uint4*)&bf[g])[1] = *(const uint4*)&Bl[buf][col][kb2 + 16];
        }
        #pragma unroll
        for (int f = 0; f < 2; ++f)
            #pragma unroll
            for (int g = 0; g < 4; ++g)
                acc[f][g] = __builtin_amdgcn_wmma_f32_16x16x32_f16(
                    false, af[f], false, bf[g], (short)0, acc[f][g], false, false);
    }

    #pragma unroll
    for (int f = 0; f < 2; ++f)
        #pragma unroll
        for (int g = 0; g < 4; ++g)
            #pragma unroll
            for (int r = 0; r < 8; ++r) {
                int m = m0 + wm * 32 + f * 16 + r + lh * 8;
                int n = n0 + wn * 64 + g * 16 + lr;
                float vv = acc[f][g][r] + bias[n];
                if (mode == 1) {
                    outF[(size_t)m * 4096 + n] = vv;
                } else {
                    int b = m >> 10, s = m & (S_ - 1);
                    int h = n >> 7,  hd = n & (HD_ - 1);
                    if (mode == 0)
                        outH[((size_t)(b * H_ + h) * S_ + s) * HD_ + hd] = (_Float16)vv;
                    else // mode 2: V transposed [B,H,HD,S]
                        outH[((size_t)(b * H_ + h) * HD_ + hd) * S_ + s] = (_Float16)vv;
                }
            }
}

// ---------------------------------------------------------------------------
// Partial rotary on first 32 channels of q and k, [B,H,S,HD], in place.
// ---------------------------------------------------------------------------
__global__ void rotary_kernel(_Float16* __restrict__ q, _Float16* __restrict__ k) {
    int idx = blockIdx.x * blockDim.x + threadIdx.x;       // B*H*S
    if (idx >= B_ * H_ * S_) return;
    int s = idx & (S_ - 1);
    float pos = (float)(10000 - S_ + s);
    const float LOG2_10000 = 13.287712379549449f;
    _Float16* bufs[2] = { q + (size_t)idx * HD_, k + (size_t)idx * HD_ };
    #pragma unroll
    for (int p = 0; p < 2; ++p) {
        _Float16* x = bufs[p];
        float re[16], im[16];
        #pragma unroll
        for (int j = 0; j < 16; ++j) { re[j] = (float)x[j]; im[j] = (float)x[16 + j]; }
        #pragma unroll
        for (int j = 0; j < 16; ++j) {
            float invf = exp2f(-LOG2_10000 * (float)(2 * j) * (1.0f / 32.0f));
            float ang  = pos * invf;
            float sv, cv;
            __sincosf(ang, &sv, &cv);
            x[2 * j]     = (_Float16)(re[j] * cv - im[j] * sv);
            x[2 * j + 1] = (_Float16)(re[j] * sv + im[j] * cv);
        }
    }
}

// ---------------------------------------------------------------------------
// Flash attention: block = one (b,h) x 64 query rows; 4 waves x 16 rows.
// K [bh][S][HD] staged row-major; V pre-transposed [bh][HD][S] staged row-major.
// Both stages are contiguous async copies, double buffered.
// ---------------------------------------------------------------------------
__global__ __launch_bounds__(128) void flash_attn_kernel(
    const _Float16* __restrict__ q,      // [B*H][S][HD]
    const _Float16* __restrict__ k,      // [B*H][S][HD]
    const _Float16* __restrict__ vT,     // [B*H][HD][S]
    const float*    __restrict__ bias,   // [S][S]
    _Float16* __restrict__ o)            // [B*S][H*HD]
{
    __shared__ _Float16 Kl[2][32][136];
    __shared__ _Float16 Vl[2][128][40];
    __shared__ _Float16 Pl[4][16][40];

    const int tid  = threadIdx.x;
    const int lane = tid & 31;
    const int wave = tid >> 5;
    const int lr   = lane & 15;
    const int lh   = lane >> 4;
    const int qt   = blockIdx.x;              // 0..15
    const int bh   = blockIdx.y;              // 0..127
    const int b    = bh >> 5, h = bh & 31;
    const int q0   = qt * 64 + wave * 16;
    const size_t kbase = (size_t)bh * S_ * HD_;
    const size_t vbase = (size_t)bh * HD_ * S_;
    const float scale = 0.08838834764831845f; // 1/sqrt(128)

    // Q fragments resident in registers for the whole key loop
    v16h qf[4];
    {
        const _Float16* qrow = q + kbase + (size_t)(q0 + lr) * HD_;
        #pragma unroll
        for (int c = 0; c < 4; ++c) {
            int hb = c * 32 + lh * 8;
            ((uint4*)&qf[c])[0] = *(const uint4*)&qrow[hb];
            ((uint4*)&qf[c])[1] = *(const uint4*)&qrow[hb + 16];
        }
    }

    v8f   ofr[8];
    float mrun[8], lrun[8];
    #pragma unroll
    for (int i = 0; i < 8; ++i) {
        #pragma unroll
        for (int e = 0; e < 8; ++e) ofr[i][e] = 0.0f;
        mrun[i] = -1e30f;
        lrun[i] = 0.0f;
    }

    auto stage = [&](int bsel, int kb0) {
        #pragma unroll
        for (int u = 0; u < 4; ++u) {
            int j = tid * 4 + u;                      // 0..511
            {   // K block: 32 keys x 128 hd
                int row = j >> 4, c8 = (j & 15) * 8;
                copy16_g2l(&k[kbase + (size_t)(kb0 + row) * HD_ + c8],
                           &Kl[bsel][row][c8]);
            }
            {   // V block: 128 hd x 32 keys (already transposed in global)
                int hd = j >> 2, c4 = (j & 3) * 8;
                copy16_g2l(&vT[vbase + (size_t)hd * S_ + kb0 + c4],
                           &Vl[bsel][hd][c4]);
            }
        }
    };

    stage(0, 0);
    for (int kt = 0, buf = 0; kt < S_ / 32; ++kt, buf ^= 1) {
        const int kb0 = kt * 32;
        async_wait_all();
        __syncthreads();
        if (kb0 + 32 < S_) stage(buf ^ 1, kb0 + 32);

        // ---- S = Q K^T ----
        v8f st[2];
        #pragma unroll
        for (int t = 0; t < 2; ++t) {
            v8f a;
            #pragma unroll
            for (int e = 0; e < 8; ++e) a[e] = 0.0f;
            int key = t * 16 + lr;
            #pragma unroll
            for (int c = 0; c < 4; ++c) {
                v16h bf;
                int hb = c * 32 + lh * 8;
                ((uint4*)&bf)[0] = *(const uint4*)&Kl[buf][key][hb];
                ((uint4*)&bf)[1] = *(const uint4*)&Kl[buf][key][hb + 16];
                a = __builtin_amdgcn_wmma_f32_16x16x32_f16(
                        false, qf[c], false, bf, (short)0, a, false, false);
            }
            st[t] = a;
        }

        // ---- scale + bias ----
        #pragma unroll
        for (int t = 0; t < 2; ++t)
            #pragma unroll
            for (int r = 0; r < 8; ++r) {
                int qg = qt * 64 + wave * 16 + r + lh * 8;
                int kg = kb0 + t * 16 + lr;
                st[t][r] = st[t][r] * scale + bias[(size_t)qg * S_ + kg];
            }

        // ---- online softmax (row reductions within 16-lane halves) ----
        float alpha[8];
        #pragma unroll
        for (int r = 0; r < 8; ++r) {
            float mb = fmaxf(st[0][r], st[1][r]);
            for (int off = 8; off >= 1; off >>= 1)
                mb = fmaxf(mb, __shfl_xor(mb, off, 32));
            float mn = fmaxf(mrun[r], mb);
            alpha[r] = __expf(mrun[r] - mn);
            mrun[r]  = mn;
            float p0 = __expf(st[0][r] - mn);
            float p1 = __expf(st[1][r] - mn);
            st[0][r] = p0; st[1][r] = p1;
            float ps = p0 + p1;
            for (int off = 8; off >= 1; off >>= 1)
                ps += __shfl_xor(ps, off, 32);
            lrun[r] = lrun[r] * alpha[r] + ps;
        }

        // ---- re-lay P (C layout -> A layout) through LDS ----
        #pragma unroll
        for (int t = 0; t < 2; ++t)
            #pragma unroll
            for (int r = 0; r < 8; ++r)
                Pl[wave][r + lh * 8][t * 16 + lr] = (_Float16)st[t][r];
        __syncthreads();
        v16h pf;
        {
            int kb = lh * 8;
            ((uint4*)&pf)[0] = *(const uint4*)&Pl[wave][lr][kb];
            ((uint4*)&pf)[1] = *(const uint4*)&Pl[wave][lr][kb + 16];
        }

        // ---- O = O*alpha + P V ----
        #pragma unroll
        for (int nf = 0; nf < 8; ++nf) {
            #pragma unroll
            for (int r = 0; r < 8; ++r) ofr[nf][r] *= alpha[r];
            v16h vf;
            int hd  = nf * 16 + lr;
            int kb2 = lh * 8;
            ((uint4*)&vf)[0] = *(const uint4*)&Vl[buf][hd][kb2];
            ((uint4*)&vf)[1] = *(const uint4*)&Vl[buf][hd][kb2 + 16];
            ofr[nf] = __builtin_amdgcn_wmma_f32_16x16x32_f16(
                          false, pf, false, vf, (short)0, ofr[nf], false, false);
        }
    }

    // ---- normalize, write O to [B*S][H*HD] f16 ----
    #pragma unroll
    for (int nf = 0; nf < 8; ++nf)
        #pragma unroll
        for (int r = 0; r < 8; ++r) {
            int qg = qt * 64 + wave * 16 + r + lh * 8;
            int hd = nf * 16 + lr;
            float val = ofr[nf][r] / lrun[r];
            o[(size_t)(b * S_ + qg) * 4096 + h * HD_ + hd] = (_Float16)val;
        }
}

// ---------------------------------------------------------------------------
// Launch
// ---------------------------------------------------------------------------
extern "C" void kernel_launch(void* const* d_in, const int* in_sizes, int n_in,
                              void* d_out, int out_size, void* d_ws, size_t ws_size,
                              hipStream_t stream) {
    const float* x  = (const float*)d_in[0];
    const float* ab = (const float*)d_in[1];
    const float* wq = (const float*)d_in[2];
    const float* bq = (const float*)d_in[3];
    const float* wk = (const float*)d_in[4];
    const float* bk = (const float*)d_in[5];
    const float* wv = (const float*)d_in[6];
    const float* bv = (const float*)d_in[7];
    const float* wo = (const float*)d_in[8];
    const float* bo = (const float*)d_in[9];
    float* out = (float*)d_out;

    _Float16* xh  = (_Float16*)d_ws;        // 9 x 32MB chunks
    _Float16* wqT = xh  + NELEM;            // weights transposed: [N][K] f16
    _Float16* wkT = wqT + NELEM;
    _Float16* wvT = wkT + NELEM;
    _Float16* woT = wvT + NELEM;
    _Float16* qh  = woT + NELEM;            // [B,H,S,HD]
    _Float16* kh  = qh  + NELEM;            // [B,H,S,HD]
    _Float16* vTh = kh  + NELEM;            // [B,H,HD,S]
    _Float16* oh  = vTh + NELEM;            // [B*S][H*HD]

    // 1) casts: x straight, weights transpose-cast to [N][K]
    int n4 = (int)(NELEM / 4);
    cast_f32_to_f16_kernel<<<dim3((n4 + 255) / 256), dim3(256), 0, stream>>>(x, xh, n4);
    dim3 tg(128, 128), tb(256);
    transpose_cast_kernel<<<tg, tb, 0, stream>>>(wq, wqT);
    transpose_cast_kernel<<<tg, tb, 0, stream>>>(wk, wkT);
    transpose_cast_kernel<<<tg, tb, 0, stream>>>(wv, wvT);
    transpose_cast_kernel<<<tg, tb, 0, stream>>>(wo, woT);

    // 2) QKV projections (WMMA, async double-buffered LDS)
    dim3 gb(256), gg(32, 32);
    gemm_wmma_kernel<<<gg, gb, 0, stream>>>(xh, wqT, bq, qh,  nullptr, 0);
    gemm_wmma_kernel<<<gg, gb, 0, stream>>>(xh, wkT, bk, kh,  nullptr, 0);
    gemm_wmma_kernel<<<gg, gb, 0, stream>>>(xh, wvT, bv, vTh, nullptr, 2);

    // 3) partial rotary on q, k
    rotary_kernel<<<dim3((B_ * H_ * S_) / 256), dim3(256), 0, stream>>>(qh, kh);

    // 4) flash attention (WMMA) -> oh
    flash_attn_kernel<<<dim3(16, 128), dim3(128), 0, stream>>>(qh, kh, vTh, ab, oh);

    // 5) output projection (WMMA) -> fp32 d_out + bias
    gemm_wmma_kernel<<<gg, gb, 0, stream>>>(oh, woT, bo, nullptr, out, 1);
}
